// DMPNNLayer_82222853915227
// MI455X (gfx1250) — compile-verified, matrix operand order checked by reading, not run on hardware
//
#include <hip/hip_runtime.h>
#include <hip/hip_bf16.h>

#define N_NODES   50000
#define N_EDGES   800000
#define D_NODE    128
#define D_EDGE    64
#define D_CAT     320      // 2*D_NODE + D_EDGE
#define H_MSG     160      // D_CAT/2
#define H_CMB     192      // 3*D_NODE/2
#define N_DIR     (2*N_EDGES)   // 1,600,000 directed edges

typedef __bf16 bf16_t;
typedef __bf16 v16bf __attribute__((ext_vector_type(16)));
typedef __bf16 v8bf  __attribute__((ext_vector_type(8)));
typedef float  v8f   __attribute__((ext_vector_type(8)));

union V16U { v16bf v; v8bf h[2]; };

// Branch-free GELU (tanh form) using the gfx1250 hardware tanh transcendental.
__device__ __forceinline__ float gelu_fast(float x) {
    float u = 0.7978845608028654f * (x + 0.044715f * x * x * x);
#if __has_builtin(__builtin_amdgcn_tanhf)
    float t = __builtin_amdgcn_tanhf(u);
#elif __has_builtin(__builtin_amdgcn_tanh_f32)
    float t = __builtin_amdgcn_tanh_f32(u);
#else
    float t = tanhf(u);
#endif
    return 0.5f * x * (1.0f + t);
}

__device__ __forceinline__ v8f wmma_bf16(v16bf a, v16bf b, v8f c) {
    // D = A(16x32 bf16) * B(32x16 bf16) + C(16x16 f32)
    return __builtin_amdgcn_wmma_f32_16x16x32_bf16(false, a, false, b, (short)0, c, false, false);
}

// Convert 8 consecutive f32 -> v8bf
__device__ __forceinline__ v8bf cvt8(const float* __restrict__ p) {
    float4 a = *(const float4*)p;
    float4 b = *(const float4*)(p + 4);
    v8bf r;
    r[0] = (bf16_t)a.x; r[1] = (bf16_t)a.y; r[2] = (bf16_t)a.z; r[3] = (bf16_t)a.w;
    r[4] = (bf16_t)b.x; r[5] = (bf16_t)b.y; r[6] = (bf16_t)b.z; r[7] = (bf16_t)b.w;
    return r;
}

// feats row = [node[dst](128) | node[src](128) | edge(64)], fetch 8 at k8 (multiple of 8)
__device__ __forceinline__ v8bf feat8(const float* d, const float* s, const float* ev, int k8) {
    const float* p = (k8 < 128) ? (d + k8) : (k8 < 256) ? (s + (k8 - 128)) : (ev + (k8 - 256));
    return cvt8(p);
}

// pairs row = [node(128) | msgbox(128)]
__device__ __forceinline__ v8bf pair8(const float* n, const float* m, int k8) {
    const float* p = (k8 < 128) ? (n + k8) : (m + (k8 - 128));
    return cvt8(p);
}

// Pre-swizzle a row-major fp32 weight (Ktot x Ntot) into bf16 WMMA-B lane layout in LDS.
// Tile (kt,nt) at (kt*numNt+nt)*512 + lane*16:
//   lane<16 : col = nt*16+lane,      K order = kt*32 + {0..7, 16..23}
//   lane>=16: col = nt*16+(lane-16), K order = kt*32 + {8..15,24..31}
__device__ void fill_Btiles(bf16_t* __restrict__ dstLds, const float* __restrict__ srcW,
                            int Ktot, int Ntot) {
    int numNt = Ntot >> 4;
    int numKt = Ktot >> 5;
    int entries = numKt * numNt * 32;
    for (int t = threadIdx.x; t < entries; t += blockDim.x) {
        int kt   = t / (numNt * 32);
        int r    = t - kt * (numNt * 32);
        int nt   = r >> 5;
        int lane = r & 31;
        int half = lane >> 4;
        int col  = nt * 16 + (lane & 15);
        int kb   = kt * 32 + half * 8;
        bf16_t* o = dstLds + t * 16;
        #pragma unroll
        for (int j = 0; j < 8; ++j) o[j]     = (bf16_t)srcW[(kb + j) * Ntot + col];
        #pragma unroll
        for (int j = 0; j < 8; ++j) o[8 + j] = (bf16_t)srcW[(kb + 16 + j) * Ntot + col];
    }
}

__device__ __forceinline__ v16bf loadB(const bf16_t* __restrict__ lds, int tile, int lane) {
    V16U t;
    const v8bf* p = (const v8bf*)(lds + tile * 512 + lane * 16);
    t.h[0] = p[0];
    t.h[1] = p[1];
    return t.v;
}

// ---------------- Kernel 1: edge MLP (M=32 per wave) + atomic scatter-add ----------------
// Each wave computes 32 directed edges per tile: two 16-row A sets share every B-tile
// LDS load, halving LDS bandwidth per WMMA.
// LDS: W1B 51200 | W2B 20480 | staging 8*(32*160)  => 225280 B
__global__ void __launch_bounds__(256) edge_msg_kernel(
    const float* __restrict__ node, const float* __restrict__ edgev,
    const int* __restrict__ eidx,
    const float* __restrict__ W1, const float* __restrict__ b1,
    const float* __restrict__ W2, const float* __restrict__ b2,
    float* __restrict__ msgbox)
{
    extern __shared__ bf16_t lds[];
    bf16_t* W1B  = lds;                      // 10 kt * 10 nt tiles
    bf16_t* W2B  = lds + 51200;              // 5 kt * 8 nt tiles
    bf16_t* stag = lds + 51200 + 20480;      // 8 waves * 32*H_MSG

    fill_Btiles(W1B, W1, D_CAT, H_MSG);
    fill_Btiles(W2B, W2, H_MSG, D_NODE);
    __syncthreads();

    const int wave = threadIdx.x >> 5;
    const int lane = threadIdx.x & 31;
    const int half = lane >> 4;
    const int lm   = lane & 15;
    bf16_t* myStag = stag + wave * (32 * H_MSG);

    // Loop-invariant per-lane biases (each lane owns column lm of every n-tile)
    float bias1[10], bias2[8];
    #pragma unroll
    for (int nt = 0; nt < 10; ++nt) bias1[nt] = b1[nt * 16 + lm];
    #pragma unroll
    for (int nt = 0; nt < 8;  ++nt) bias2[nt] = b2[nt * 16 + lm];

    const int nTiles = N_DIR / 32;           // 50000 (N_EDGES%32==0 -> halves uniform)
    for (int tile = blockIdx.x * 8 + wave; tile < nTiles; tile += gridDim.x * 8) {
        const int ebase = tile * 32;

        // Lane lm gathers feats for rows lm (set a) and 16+lm (set b).
        int ea = ebase + lm;
        int eb = ebase + 16 + lm;
        int eua = (ea < N_EDGES) ? ea : ea - N_EDGES;
        int eub = (eb < N_EDGES) ? eb : eb - N_EDGES;
        int a0 = eidx[2 * eua], a1 = eidx[2 * eua + 1];
        int b0 = eidx[2 * eub], b1i = eidx[2 * eub + 1];
        int dna = (ea < N_EDGES) ? a0 : a1;
        int sna = (ea < N_EDGES) ? a1 : a0;
        int dnb = (eb < N_EDGES) ? b0 : b1i;
        int snb = (eb < N_EDGES) ? b1i : b0;
        const float* dpa = node + (long)dna * D_NODE;
        const float* spa = node + (long)sna * D_NODE;
        const float* epa = edgev + (long)eua * D_EDGE;
        const float* dpb = node + (long)dnb * D_NODE;
        const float* spb = node + (long)snb * D_NODE;
        const float* epb = edgev + (long)eub * D_EDGE;

        // Build two A-register sets (rows 0-15 and 16-31)
        v16bf Aa[10], Ab[10];
        #pragma unroll
        for (int kt = 0; kt < 10; ++kt) {
            int kb = kt * 32 + half * 8;
            V16U ta, tb;
            ta.h[0] = feat8(dpa, spa, epa, kb);
            ta.h[1] = feat8(dpa, spa, epa, kb + 16);
            tb.h[0] = feat8(dpb, spb, epb, kb);
            tb.h[1] = feat8(dpb, spb, epb, kb + 16);
            Aa[kt] = ta.v;
            Ab[kt] = tb.v;
        }

        // h = gelu(feats @ W1 + b1): 32x160, stage to LDS. B double-buffered, reused x2.
        #pragma unroll
        for (int nt = 0; nt < 10; ++nt) {
            v8f ca = {}, cb = {};
            v16bf Bcur = loadB(W1B, nt, lane);
            #pragma unroll
            for (int kt = 0; kt < 10; ++kt) {
                v16bf Bnext = Bcur;
                if (kt < 9) Bnext = loadB(W1B, (kt + 1) * 10 + nt, lane);
                ca = wmma_bf16(Aa[kt], Bcur, ca);
                cb = wmma_bf16(Ab[kt], Bcur, cb);
                Bcur = Bnext;
            }
            float bias = bias1[nt];
            #pragma unroll
            for (int r = 0; r < 8; ++r) {
                int m = half * 8 + r;                       // D row held by this lane
                myStag[m * H_MSG + nt * 16 + lm]        = (bf16_t)gelu_fast(ca[r] + bias);
                myStag[(16 + m) * H_MSG + nt * 16 + lm] = (bf16_t)gelu_fast(cb[r] + bias);
            }
        }

        __asm volatile("s_wait_dscnt 0" ::: "memory");      // same-wave DS RAW ordering

        // Reload h as WMMA A-layout (transpose via LDS)
        v16bf A2a[5], A2b[5];
        #pragma unroll
        for (int kt = 0; kt < 5; ++kt) {
            int kb = kt * 32 + half * 8;
            V16U ta, tb;
            ta.h[0] = *(const v8bf*)(myStag + lm * H_MSG + kb);
            ta.h[1] = *(const v8bf*)(myStag + lm * H_MSG + kb + 16);
            tb.h[0] = *(const v8bf*)(myStag + (16 + lm) * H_MSG + kb);
            tb.h[1] = *(const v8bf*)(myStag + (16 + lm) * H_MSG + kb + 16);
            A2a[kt] = ta.v;
            A2b[kt] = tb.v;
        }

        // Destination node per D-row: lane m already holds it; broadcast via shfl.
        int dstma[8], dstmb[8];
        #pragma unroll
        for (int r = 0; r < 8; ++r) {
            dstma[r] = __shfl(dna, half * 8 + r, 32);
            dstmb[r] = __shfl(dnb, half * 8 + r, 32);
        }

        // messages = gelu(h @ W2 + b2), scatter-add into msgbox (segment_sum)
        #pragma unroll
        for (int nt = 0; nt < 8; ++nt) {
            v8f ca = {}, cb = {};
            v16bf Bcur = loadB(W2B, nt, lane);
            #pragma unroll
            for (int kt = 0; kt < 5; ++kt) {
                v16bf Bnext = Bcur;
                if (kt < 4) Bnext = loadB(W2B, (kt + 1) * 8 + nt, lane);
                ca = wmma_bf16(A2a[kt], Bcur, ca);
                cb = wmma_bf16(A2b[kt], Bcur, cb);
                Bcur = Bnext;
            }
            float bias = bias2[nt];
            int col = nt * 16 + lm;
            #pragma unroll
            for (int r = 0; r < 8; ++r) {
                atomicAdd(&msgbox[(long)dstma[r] * D_NODE + col], gelu_fast(ca[r] + bias));
                atomicAdd(&msgbox[(long)dstmb[r] * D_NODE + col], gelu_fast(cb[r] + bias));
            }
        }
    }
}

// ---------------- Kernel 2: node combine MLP (M=16 per wave) -----------------------------
// LDS: V1B 49152 | V2B 24576 | staging 8*(16*192)  => 196608 B
__global__ void __launch_bounds__(256) node_combine_kernel(
    const float* __restrict__ node, const float* __restrict__ msgbox,
    const float* __restrict__ V1, const float* __restrict__ c1,
    const float* __restrict__ V2, const float* __restrict__ c2,
    float* __restrict__ out)
{
    extern __shared__ bf16_t lds[];
    bf16_t* V1B  = lds;                      // 8 kt * 12 nt
    bf16_t* V2B  = lds + 49152;              // 6 kt * 8 nt
    bf16_t* stag = lds + 49152 + 24576;

    fill_Btiles(V1B, V1, 2 * D_NODE, H_CMB);
    fill_Btiles(V2B, V2, H_CMB, D_NODE);
    __syncthreads();

    const int wave = threadIdx.x >> 5;
    const int lane = threadIdx.x & 31;
    const int half = lane >> 4;
    const int lm   = lane & 15;
    bf16_t* myStag = stag + wave * (16 * H_CMB);

    float bias1[12], bias2[8];
    #pragma unroll
    for (int nt = 0; nt < 12; ++nt) bias1[nt] = c1[nt * 16 + lm];
    #pragma unroll
    for (int nt = 0; nt < 8;  ++nt) bias2[nt] = c2[nt * 16 + lm];

    const int nTiles = N_NODES / 16;         // 3125
    for (int tile = blockIdx.x * 8 + wave; tile < nTiles; tile += gridDim.x * 8) {
        const int rbase = tile * 16;
        const float* nptr = node   + (long)(rbase + lm) * D_NODE;
        const float* mptr = msgbox + (long)(rbase + lm) * D_NODE;

        v16bf A[8];
        #pragma unroll
        for (int kt = 0; kt < 8; ++kt) {
            int kb = kt * 32 + half * 8;
            V16U t;
            t.h[0] = pair8(nptr, mptr, kb);
            t.h[1] = pair8(nptr, mptr, kb + 16);
            A[kt] = t.v;
        }

        // g = gelu(pairs @ V1 + c1): 16x192, stage to LDS
        #pragma unroll
        for (int nt = 0; nt < 12; ++nt) {
            v8f c = {};
            v16bf Bcur = loadB(V1B, nt, lane);
            #pragma unroll
            for (int kt = 0; kt < 8; ++kt) {
                v16bf Bnext = Bcur;
                if (kt < 7) Bnext = loadB(V1B, (kt + 1) * 12 + nt, lane);
                c = wmma_bf16(A[kt], Bcur, c);
                Bcur = Bnext;
            }
            float bias = bias1[nt];
            #pragma unroll
            for (int r = 0; r < 8; ++r) {
                int m = half * 8 + r;
                myStag[m * H_CMB + nt * 16 + lm] = (bf16_t)gelu_fast(c[r] + bias);
            }
        }

        __asm volatile("s_wait_dscnt 0" ::: "memory");

        v16bf A2[6];
        #pragma unroll
        for (int kt = 0; kt < 6; ++kt) {
            int kb = kt * 32 + half * 8;
            V16U t;
            t.h[0] = *(const v8bf*)(myStag + lm * H_CMB + kb);
            t.h[1] = *(const v8bf*)(myStag + lm * H_CMB + kb + 16);
            A2[kt] = t.v;
        }

        // out = gelu(g @ V2 + c2)
        #pragma unroll
        for (int nt = 0; nt < 8; ++nt) {
            v8f c = {};
            v16bf Bcur = loadB(V2B, nt, lane);
            #pragma unroll
            for (int kt = 0; kt < 6; ++kt) {
                v16bf Bnext = Bcur;
                if (kt < 5) Bnext = loadB(V2B, (kt + 1) * 8 + nt, lane);
                c = wmma_bf16(A2[kt], Bcur, c);
                Bcur = Bnext;
            }
            float bias = bias2[nt];
            int col = nt * 16 + lm;
            #pragma unroll
            for (int r = 0; r < 8; ++r)
                out[(long)(rbase + half * 8 + r) * D_NODE + col] = gelu_fast(c[r] + bias);
        }
    }
}

extern "C" void kernel_launch(void* const* d_in, const int* in_sizes, int n_in,
                              void* d_out, int out_size, void* d_ws, size_t ws_size,
                              hipStream_t stream)
{
    const float* node  = (const float*)d_in[0];
    const float* edgev = (const float*)d_in[1];
    const int*   eidx  = (const int*)  d_in[2];
    const float* W1    = (const float*)d_in[3];
    const float* b1    = (const float*)d_in[4];
    const float* W2    = (const float*)d_in[5];
    const float* b2    = (const float*)d_in[6];
    const float* V1    = (const float*)d_in[7];
    const float* c1    = (const float*)d_in[8];
    const float* V2    = (const float*)d_in[9];
    const float* c2    = (const float*)d_in[10];
    float* out    = (float*)d_out;
    float* msgbox = (float*)d_ws;

    // zero the segment-sum accumulator (graph-capture-safe)
    hipMemsetAsync(msgbox, 0, (size_t)N_NODES * D_NODE * sizeof(float), stream);

    size_t lds1 = (size_t)(51200 + 20480 + 8 * 32 * H_MSG) * sizeof(bf16_t);  // 225280 B
    size_t lds2 = (size_t)(49152 + 24576 + 8 * 16 * H_CMB) * sizeof(bf16_t);  // 196608 B
    hipFuncSetAttribute((const void*)edge_msg_kernel,
                        hipFuncAttributeMaxDynamicSharedMemorySize, (int)lds1);
    hipFuncSetAttribute((const void*)node_combine_kernel,
                        hipFuncAttributeMaxDynamicSharedMemorySize, (int)lds2);

    // Kernel 1: 50000 thirty-two-edge wave-tiles, 8 waves/block, grid-stride
    edge_msg_kernel<<<dim3(1250), dim3(256), lds1, stream>>>(
        node, edgev, eidx, W1, b1, W2, b2, msgbox);

    // Kernel 2: 3125 node tiles
    node_combine_kernel<<<dim3(391), dim3(256), lds2, stream>>>(
        node, msgbox, V1, c1, V2, c2, out);
}
